// TimeAttention_37306085933193
// MI455X (gfx1250) — compile-verified, matrix operand order checked by reading, not run on hardware
//
#include <hip/hip_runtime.h>

// ---------------------------------------------------------------------------
// MI455X (gfx1250) fused attention: QKV+RoPE -> softmax attention -> out proj
// All matrix math via v_wmma_f32_16x16x32_f16 (wave32, f32 accumulate).
// Memory strategy: all fragment traffic is wide/contiguous f16:
//   - hidden pre-converted & permuted to f16 (kernel 0a)
//   - weights pre-packed into WMMA B-fragment order (kernel 0b)
//   - V written transposed so attention B-frags are contiguous
// ---------------------------------------------------------------------------

typedef __attribute__((ext_vector_type(16))) _Float16 v16h;
typedef __attribute__((ext_vector_type(8)))  _Float16 v8h;
typedef __attribute__((ext_vector_type(8)))  float    v8f;
typedef __attribute__((ext_vector_type(4)))  float    v4f;

#define D_MODEL   1024
#define N_HEADS   16
#define N_KVH     4
#define HEAD_DIM  64
#define SEQ       512
#define BP        16          // B*P = 2*8
#define ROWS      8192        // BP * SEQ
#define KCH       (D_MODEL / 32)   // 32 k-chunks of 32

__device__ __forceinline__ v8f wmma16x16x32(v16h a, v16h b, v8f c) {
  // (neg_a, A, neg_b, B, c_mod, C, reuse_a, reuse_b)
  return __builtin_amdgcn_wmma_f32_16x16x32_f16(false, a, false, b, (short)0, c,
                                                false, false);
}

__device__ __forceinline__ v16h cat8(v8h lo, v8h hi) {
  v16h r;
#pragma unroll
  for (int i = 0; i < 8; ++i) { r[i] = lo[i]; r[i + 8] = hi[i]; }
  return r;
}

// ===========================================================================
// Kernel 0a: hidden (b,t,p,d) fp32 -> xh (b,p,t,d) f16.  Fully coalesced.
// One thread handles 8 consecutive d.  grid = ROWS*D_MODEL/8/256 blocks.
// ===========================================================================
__global__ __launch_bounds__(256)
void convert_x_kernel(const float* __restrict__ hidden,
                      _Float16* __restrict__ xh)
{
  const int idx = blockIdx.x * 256 + threadIdx.x;     // 0 .. 1M-1
  const int d0  = (idx & 127) << 3;                   // 8-elem group in row
  const int rr  = idx >> 7;                           // (b,p,t) row
  const int bb  = rr >> 12;
  const int pp  = (rr >> 9) & 7;
  const int tt  = rr & (SEQ - 1);
  const float* src = hidden + (((size_t)(bb * SEQ + tt) * 8 + pp) << 10) + d0;
  v4f f0 = *(const v4f*)(src);
  v4f f1 = *(const v4f*)(src + 4);
  v8h h;
#pragma unroll
  for (int i = 0; i < 4; ++i) {
    h[i]     = (_Float16)f0[i];
    h[i + 4] = (_Float16)f1[i];
  }
  *(v8h*)(xh + (size_t)rr * D_MODEL + d0) = h;
}

// ===========================================================================
// Kernel 0b: pack fp32 weight (K=1024 x N) into f16 B-fragment order:
//   dst[((nt*KCH + kc)*32 + lane)*16 + i] = W[k(lane,i)*N + nt*16 + (lane&15)]
// with k = kc*32 + koff + i (i<8) / kc*32 + 16 + koff + i-8, koff=(lane>>4)*8.
// grid = (N/16, KCH), 32 threads. Runs once; packed weights stay in L2.
// ===========================================================================
__global__ __launch_bounds__(32)
void pack_w_kernel(const float* __restrict__ W, int N,
                   _Float16* __restrict__ dst)
{
  const int lane = threadIdx.x;
  const int ln   = lane & 15;
  const int koff = (lane >> 4) << 3;
  const int nt   = blockIdx.x;
  const int kc   = blockIdx.y;
  const int kb   = kc * 32;
  const float* col = W + nt * 16 + ln;
  v16h b;
#pragma unroll
  for (int i = 0; i < 8; ++i) {
    b[i]     = (_Float16)col[(size_t)(kb + koff + i) * N];
    b[i + 8] = (_Float16)col[(size_t)(kb + 16 + koff + i) * N];
  }
  *(v16h*)(dst + ((size_t)(nt * KCH + kc) * 32 + lane) * 16) = b;
}

// ===========================================================================
// Kernel 1: QKV projection + RoPE + q-scale from packed f16 operands.
// 4 waves / block, wave w handles row tile rt = blockIdx.x*4 + w (16 rows)
// for one 64-wide head column (blockIdx.y in [0,24)).
// Q/K written row-major (slot,t,d) via LDS-staged wide stores.
// V written TRANSPOSED (slot,d,t) straight from registers (wide stores).
// ===========================================================================
__global__ __launch_bounds__(128)
void qkv_rope_kernel(const _Float16* __restrict__ xh,
                     const _Float16* __restrict__ pWq,
                     const _Float16* __restrict__ pWk,
                     const _Float16* __restrict__ pWv,
                     _Float16* __restrict__ qws,
                     _Float16* __restrict__ kws,
                     _Float16* __restrict__ vT)
{
  __shared__ __align__(128) _Float16 st[4][16][64];   // 8 KB staging

  const int lane = threadIdx.x & 31;
  const int w    = threadIdx.x >> 5;
  const int ln   = lane & 15;
  const int koff = (lane >> 4) << 3;
  const int mb   = (lane >> 4) << 3;
  const int rt   = blockIdx.x * 4 + w;
  const int hc   = blockIdx.y;

  const _Float16* pW; int ntbase, mode, head;
  if (hc < 16)      { pW = pWq; ntbase = hc * 4;        mode = 0; head = hc; }
  else if (hc < 20) { pW = pWk; ntbase = (hc - 16) * 4; mode = 1; head = hc - 16; }
  else              { pW = pWv; ntbase = (hc - 20) * 4; mode = 2; head = hc - 20; }

  const _Float16* __restrict__ xrow = xh + (size_t)(rt * 16 + ln) * D_MODEL;

  v8f acc0 = {}, acc1 = {}, acc2 = {}, acc3 = {};
  for (int kc = 0; kc < KCH; ++kc) {
    const int kb = kc * 32;
    v16h a = cat8(*(const v8h*)(xrow + kb + koff),
                  *(const v8h*)(xrow + kb + 16 + koff));
    const _Float16* pb = pW + ((size_t)kc * 32 + lane) * 16;
    const size_t ntstride = (size_t)KCH * 32 * 16;
    v16h b0 = *(const v16h*)(pb + (size_t)(ntbase + 0) * ntstride);
    v16h b1 = *(const v16h*)(pb + (size_t)(ntbase + 1) * ntstride);
    v16h b2 = *(const v16h*)(pb + (size_t)(ntbase + 2) * ntstride);
    v16h b3 = *(const v16h*)(pb + (size_t)(ntbase + 3) * ntstride);
    acc0 = wmma16x16x32(a, b0, acc0);
    acc1 = wmma16x16x32(a, b1, acc1);
    acc2 = wmma16x16x32(a, b2, acc2);
    acc3 = wmma16x16x32(a, b3, acc3);
  }

  const int t0 = (rt * 16) & (SEQ - 1);
  const int bp = rt >> 5;

  if (mode < 2) {
    // RoPE pairs (dh, dh+32): acc0<->acc2 (dh=ln), acc1<->acc3 (dh=16+ln)
    const float LOG1E4_32 = 0.28782313662425575f;    // ln(10000)/32
    const float fA = __expf(-(float)ln * LOG1E4_32);
    const float fB = __expf(-(float)(16 + ln) * LOG1E4_32);
    const float qs = (mode == 0) ? 0.125f : 1.0f;    // 1/sqrt(HEAD_DIM) into q
#pragma unroll
    for (int r = 0; r < 8; ++r) {
      const float tm = (float)(t0 + mb + r);
      float sA, cA, sB, cB;
      __sincosf(tm * fA, &sA, &cA);
      __sincosf(tm * fB, &sB, &cB);
      float lo = acc0[r], hi = acc2[r];
      acc0[r] = (lo * cA - hi * sA) * qs;
      acc2[r] = (lo * sA + hi * cA) * qs;
      lo = acc1[r]; hi = acc3[r];
      acc1[r] = (lo * cB - hi * sB) * qs;
      acc3[r] = (lo * sB + hi * cB) * qs;
    }
    // stage tile in LDS, then 2x32B stores per lane
    _Float16* dst = (mode == 0)
        ? qws + ((size_t)(bp * N_HEADS + head) * SEQ) * HEAD_DIM
        : kws + ((size_t)(bp * N_KVH + head) * SEQ) * HEAD_DIM;
#pragma unroll
    for (int r = 0; r < 8; ++r) {
      _Float16* srow = &st[w][mb + r][0];
      srow[ln]      = (_Float16)acc0[r];
      srow[16 + ln] = (_Float16)acc1[r];
      srow[32 + ln] = (_Float16)acc2[r];
      srow[48 + ln] = (_Float16)acc3[r];
    }
    __syncthreads();
    const int seg = lane >> 4;
    const v16h* s = (const v16h*)&st[w][ln][seg * 32];
    v16h* g = (v16h*)(dst + (size_t)(t0 + ln) * HEAD_DIM + seg * 32);
    g[0] = s[0];
    g[1] = s[1];
  } else {
    // V: transpose in-register -> vT[slot][d][t], 4 x 16B stores per lane
    _Float16* dst = vT + ((size_t)(bp * N_KVH + head) * HEAD_DIM) * SEQ;
#pragma unroll
    for (int j = 0; j < 4; ++j) {
      v8f* accs[4] = {&acc0, &acc1, &acc2, &acc3};
      v8f a = *accs[j];
      v8h h;
#pragma unroll
      for (int r = 0; r < 8; ++r) h[r] = (_Float16)a[r];
      *(v8h*)(dst + (size_t)(j * 16 + ln) * SEQ + t0 + mb) = h;
    }
    __syncthreads();   // keep block convergent with mode<2 path (same hc => same path; harmless)
  }
}

// ===========================================================================
// Kernel 2: attention, one wave per (bp, head, 16-query tile).
//   grid = (SEQ/16, N_HEADS, BP)
// f16 scores/probs in one 16 KB LDS buffer, exact softmax in fp32 registers,
// AV via WMMA with contiguous V^T fragments, ctx stored via LDS-staged 32B.
// ===========================================================================
__global__ __launch_bounds__(32)
void attn_kernel(const _Float16* __restrict__ qws,
                 const _Float16* __restrict__ kws,
                 const _Float16* __restrict__ vT,
                 _Float16* __restrict__ ctx)
{
  __shared__ __align__(128) _Float16 sc[16][SEQ];     // 16 KB scores/probs

  const int lane = threadIdx.x;
  const int ln   = lane & 15;
  const int koff = (lane >> 4) << 3;
  const int mb   = (lane >> 4) << 3;
  const int qt   = blockIdx.x;
  const int h    = blockIdx.y;
  const int bp   = blockIdx.z;

  const _Float16* __restrict__ qb = qws + ((size_t)(bp * N_HEADS + h)        * SEQ) * HEAD_DIM;
  const _Float16* __restrict__ kb = kws + ((size_t)(bp * N_KVH   + (h >> 2)) * SEQ) * HEAD_DIM;
  const _Float16* __restrict__ vb = vT  + ((size_t)(bp * N_KVH   + (h >> 2)) * HEAD_DIM) * SEQ;

  // Q tile A-fragments (16x64 as two 16x32 chunks)
  const _Float16* qrow = qb + (size_t)(qt * 16 + ln) * HEAD_DIM;
  v16h aq0 = cat8(*(const v8h*)(qrow + koff),      *(const v8h*)(qrow + 16 + koff));
  v16h aq1 = cat8(*(const v8h*)(qrow + 32 + koff), *(const v8h*)(qrow + 48 + koff));

  // S = Q K^T : per-lane key row is contiguous, identical gather as A
  for (int kt = 0; kt < SEQ / 16; ++kt) {
    const _Float16* krow = kb + (size_t)(kt * 16 + ln) * HEAD_DIM;
    if (kt + 1 < SEQ / 16)
      __builtin_prefetch(krow + HEAD_DIM * 16, 0, 1);   // global_prefetch_b8
    v16h bk0 = cat8(*(const v8h*)(krow + koff),      *(const v8h*)(krow + 16 + koff));
    v16h bk1 = cat8(*(const v8h*)(krow + 32 + koff), *(const v8h*)(krow + 48 + koff));
    v8f c = {};
    c = wmma16x16x32(aq0, bk0, c);
    c = wmma16x16x32(aq1, bk1, c);
    const int col = kt * 16 + ln;
#pragma unroll
    for (int r = 0; r < 8; ++r) sc[mb + r][col] = (_Float16)c[r];
  }
  __syncthreads();

  // Exact softmax over 512 keys (fp32 registers, full-wave shuffle reductions)
  for (int row = 0; row < 16; ++row) {
    float vals[16];
    float mx = -3.0e38f;
#pragma unroll
    for (int j = 0; j < 16; ++j) {
      vals[j] = (float)sc[row][lane + 32 * j];
      mx = fmaxf(mx, vals[j]);
    }
#pragma unroll
    for (int off = 16; off > 0; off >>= 1)
      mx = fmaxf(mx, __shfl_xor(mx, off, 32));
    float sum = 0.f;
#pragma unroll
    for (int j = 0; j < 16; ++j) {
      vals[j] = __expf(vals[j] - mx);
      sum += vals[j];
    }
#pragma unroll
    for (int off = 16; off > 0; off >>= 1)
      sum += __shfl_xor(sum, off, 32);
    const float rinv = 1.0f / sum;
#pragma unroll
    for (int j = 0; j < 16; ++j)
      sc[row][lane + 32 * j] = (_Float16)(vals[j] * rinv);
  }
  __syncthreads();

  // O = P V : P rows contiguous in LDS, V^T rows contiguous in global
  v8f o0 = {}, o1 = {}, o2 = {}, o3 = {};
  const _Float16* prow = &sc[ln][0];
  const _Float16* v0r = vb + (size_t)(ln)      * SEQ;
  const _Float16* v1r = vb + (size_t)(16 + ln) * SEQ;
  const _Float16* v2r = vb + (size_t)(32 + ln) * SEQ;
  const _Float16* v3r = vb + (size_t)(48 + ln) * SEQ;
  for (int kt = 0; kt < SEQ / 32; ++kt) {
    const int kbse = kt * 32;
    v16h ap  = cat8(*(const v8h*)(prow + kbse + koff), *(const v8h*)(prow + kbse + 16 + koff));
    v16h bv0 = cat8(*(const v8h*)(v0r + kbse + koff),  *(const v8h*)(v0r + kbse + 16 + koff));
    v16h bv1 = cat8(*(const v8h*)(v1r + kbse + koff),  *(const v8h*)(v1r + kbse + 16 + koff));
    v16h bv2 = cat8(*(const v8h*)(v2r + kbse + koff),  *(const v8h*)(v2r + kbse + 16 + koff));
    v16h bv3 = cat8(*(const v8h*)(v3r + kbse + koff),  *(const v8h*)(v3r + kbse + 16 + koff));
    o0 = wmma16x16x32(ap, bv0, o0);
    o1 = wmma16x16x32(ap, bv1, o1);
    o2 = wmma16x16x32(ap, bv2, o2);
    o3 = wmma16x16x32(ap, bv3, o3);
  }

  // stage 16x64 out tile into LDS (reuse sc), then 2x32B stores per lane
  __syncthreads();
  _Float16 (*ot)[64] = (_Float16(*)[64])sc;
#pragma unroll
  for (int r = 0; r < 8; ++r) {
    _Float16* srow = &ot[mb + r][0];
    srow[ln]      = (_Float16)o0[r];
    srow[16 + ln] = (_Float16)o1[r];
    srow[32 + ln] = (_Float16)o2[r];
    srow[48 + ln] = (_Float16)o3[r];
  }
  __syncthreads();
  const int seg = lane >> 4;
  const v16h* s = (const v16h*)&ot[ln][seg * 32];
  _Float16* grow = ctx + ((size_t)bp * SEQ + qt * 16 + ln) * D_MODEL
                       + h * HEAD_DIM + seg * 32;
  ((v16h*)grow)[0] = s[0];
  ((v16h*)grow)[1] = s[1];
}

// ===========================================================================
// Kernel 3: out = ctx @ Wo (packed) with (B,T,P,:) permuted fp32 store.
// 4 waves / block; wave w: row tile blockIdx.x*4+w, col tile blockIdx.y.
// ===========================================================================
__global__ __launch_bounds__(128)
void out_proj_kernel(const _Float16* __restrict__ ctx,
                     const _Float16* __restrict__ pWo,
                     float* __restrict__ out)
{
  __shared__ __align__(128) float ot[4][16][16];      // 4 KB staging

  const int lane = threadIdx.x & 31;
  const int w    = threadIdx.x >> 5;
  const int ln   = lane & 15;
  const int koff = (lane >> 4) << 3;
  const int mb   = (lane >> 4) << 3;
  const int rt   = blockIdx.x * 4 + w;
  const int ct   = blockIdx.y;

  const _Float16* __restrict__ arow = ctx + (size_t)(rt * 16 + ln) * D_MODEL;
  const _Float16* __restrict__ pb0  = pWo + ((size_t)ct * KCH * 32 + lane) * 16;

  v8f c = {};
  for (int kc = 0; kc < KCH; ++kc) {
    const int kbse = kc * 32;
    v16h a = cat8(*(const v8h*)(arow + kbse + koff),
                  *(const v8h*)(arow + kbse + 16 + koff));
    v16h b = *(const v16h*)(pb0 + (size_t)kc * 32 * 16);
    c = wmma16x16x32(a, b, c);
  }

#pragma unroll
  for (int r = 0; r < 8; ++r) ot[w][mb + r][ln] = c[r];
  __syncthreads();

  // row = ln handled by two lanes (8 floats each, 2x16B stores)
  const int seg = lane >> 4;
  const int rr  = rt * 16 + ln;          // (b,p,t) ordered row
  const int bb  = rr >> 12;
  const int pp  = (rr >> 9) & 7;
  const int tt  = rr & (SEQ - 1);
  float* grow = out + (((size_t)(bb * SEQ + tt) * 8 + pp) << 10)
                    + ct * 16 + seg * 8;
  const v4f* s = (const v4f*)&ot[w][ln][seg * 8];
  ((v4f*)grow)[0] = s[0];
  ((v4f*)grow)[1] = s[1];
}

// ===========================================================================
extern "C" void kernel_launch(void* const* d_in, const int* in_sizes, int n_in,
                              void* d_out, int out_size, void* d_ws, size_t ws_size,
                              hipStream_t stream) {
  const float* hidden = (const float*)d_in[0];
  const float* Wq     = (const float*)d_in[1];
  const float* Wk     = (const float*)d_in[2];
  const float* Wv     = (const float*)d_in[3];
  const float* Wo     = (const float*)d_in[4];
  float*       out    = (float*)d_out;

  // f16 workspace layout (halfs). ctx aliases xh (xh dead after kernel 1).
  _Float16* xh  = (_Float16*)d_ws;                         // 8192*1024
  _Float16* ctx = xh;                                      // alias
  _Float16* qws = xh  + (size_t)ROWS * D_MODEL;            // 16*16*512*64
  _Float16* kws = qws + (size_t)BP * N_HEADS * SEQ * HEAD_DIM;
  _Float16* vT  = kws + (size_t)BP * N_KVH * SEQ * HEAD_DIM;
  _Float16* pWq = vT  + (size_t)BP * N_KVH * SEQ * HEAD_DIM;
  _Float16* pWk = pWq + (size_t)D_MODEL * (N_HEADS * HEAD_DIM);
  _Float16* pWv = pWk + (size_t)D_MODEL * (N_KVH * HEAD_DIM);
  _Float16* pWo = pWv + (size_t)D_MODEL * (N_KVH * HEAD_DIM);

  convert_x_kernel<<<ROWS * D_MODEL / 8 / 256, 256, 0, stream>>>(hidden, xh);
  pack_w_kernel<<<dim3(64, KCH), 32, 0, stream>>>(Wq, 1024, pWq);
  pack_w_kernel<<<dim3(16, KCH), 32, 0, stream>>>(Wk, 256,  pWk);
  pack_w_kernel<<<dim3(16, KCH), 32, 0, stream>>>(Wv, 256,  pWv);
  pack_w_kernel<<<dim3(64, KCH), 32, 0, stream>>>(Wo, 1024, pWo);

  qkv_rope_kernel<<<dim3(ROWS / 16 / 4, N_HEADS + 2 * N_KVH), 128, 0, stream>>>(
      xh, pWq, pWk, pWv, qws, kws, vT);

  attn_kernel<<<dim3(SEQ / 16, N_HEADS, BP), 32, 0, stream>>>(
      qws, kws, vT, ctx);

  out_proj_kernel<<<dim3(ROWS / 16 / 4, D_MODEL / 16), 128, 0, stream>>>(
      ctx, pWo, out);
}
